// MultiScaleSetAbstractionModule_87050397155549
// MI455X (gfx1250) — matrix operand order repaired and checked by reading, not compiled, not run on hardware
//
#include <hip/hip_runtime.h>
#include <hip/hip_bf16.h>

// ---------------------------------------------------------------------------
// MultiScaleSetAbstraction for MI455X (gfx1250, wave32, WMMA)
//   B=4, N=2048, C=64, M=512; scales: (S=32,r=0.2) and (S=64,r=0.4)
//   MLPs: 67->64->64->128 and 67->128->128->256, f16 WMMA w/ f32 accumulate
// ---------------------------------------------------------------------------

#define NPTS  2048
#define BATCH 4
#define CFEAT 64
#define MCENT 512

typedef __attribute__((ext_vector_type(16))) _Float16 v16h;
typedef __attribute__((ext_vector_type(8)))  _Float16 v8h;
typedef __attribute__((ext_vector_type(8)))  float    v8f;

// ============================ FPS ==========================================
// One block per batch. 256 threads, 8 points each held in registers.
// Matches jax scan: idx[0]=0; each step: dist=min(dist,d2(cur)); cur=argmax
// with first-index tie-break (strict > locally, min-index on tie globally).
__global__ __launch_bounds__(256) void fps_kernel(
    const float* __restrict__ pts,   // (B,3,N)
    float* __restrict__ cent,        // scratch (B,M,3)
    float* __restrict__ out_cent)    // output  (B,3,M)
{
  const int b   = blockIdx.x;
  const int tid = threadIdx.x;
  const float* px = pts + (size_t)b * 3 * NPTS;
  const float* py = px + NPTS;
  const float* pz = px + 2 * NPTS;

  float lx[8], ly[8], lz[8], dist[8];
#pragma unroll
  for (int p = 0; p < 8; ++p) {
    int j = tid * 8 + p;
    lx[p] = px[j]; ly[p] = py[j]; lz[p] = pz[j];
    dist[p] = 1e10f;
  }

  __shared__ float sd[256];
  __shared__ int   si[256];

  int cur = 0;
  for (int it = 0; it < MCENT; ++it) {
    const float cx = px[cur], cy = py[cur], cz = pz[cur];
    if (tid == 0) {
      float* c = cent + ((size_t)b * MCENT + it) * 3;
      c[0] = cx; c[1] = cy; c[2] = cz;
      out_cent[(size_t)b * 3 * MCENT + 0 * MCENT + it] = cx;
      out_cent[(size_t)b * 3 * MCENT + 1 * MCENT + it] = cy;
      out_cent[(size_t)b * 3 * MCENT + 2 * MCENT + it] = cz;
    }
    float bv = -1.0f; int bi = 0;
#pragma unroll
    for (int p = 0; p < 8; ++p) {
      float dx = lx[p] - cx, dy = ly[p] - cy, dz = lz[p] - cz;
      float d2 = __fadd_rn(__fadd_rn(__fmul_rn(dx, dx), __fmul_rn(dy, dy)),
                           __fmul_rn(dz, dz));
      float d = fminf(dist[p], d2);
      dist[p] = d;
      if (d > bv) { bv = d; bi = tid * 8 + p; }   // strict >: first max in range
    }
    sd[tid] = bv; si[tid] = bi;
    __syncthreads();
    for (int off = 128; off > 0; off >>= 1) {
      if (tid < off) {
        float ov = sd[tid + off]; int oi = si[tid + off];
        if (ov > sd[tid] || (ov == sd[tid] && oi < si[tid])) {
          sd[tid] = ov; si[tid] = oi;
        }
      }
      __syncthreads();
    }
    cur = si[0];
    __syncthreads();
  }
}

// ====================== Weight prep (f32 -> f16, transpose + K-pad) ========
// W: (cout, cin) row-major f32.  Wt: (kp, cout) f16, rows k>=cin zeroed.
// This layout puts each k-row of a WMMA B-fragment in 16 contiguous halves across lanes.
__global__ void prep_w_kernel(const float* __restrict__ W,
                              _Float16* __restrict__ Wt,
                              int cout, int cin, int kp)
{
  int n = kp * cout;
  for (int e = blockIdx.x * blockDim.x + threadIdx.x; e < n;
       e += gridDim.x * blockDim.x) {
    int k = e / cout, o = e - k * cout;
    float v = (k < cin) ? W[o * cin + k] : 0.0f;
    Wt[e] = (_Float16)v;
  }
}

// ====================== One MLP layer via v_wmma_f32_16x16x32_f16 ==========
// Xin: LDS (S x KP) f16, row stride KP (KP % 32 == 0, zero-padded K).
// Wt:  global (KP x COUT) f16.  D = relu(Xin @ Wt + bias).
// FINAL == false : write f16 activations to LDS Xout (row stride COUT).
// FINAL == true  : fold neighbor-max per column into per-wave LDS slot
//                  pmax[wid*COUT + col]; no atomics needed because each wave
//                  visits a given column group ni at most once (tile index
//                  advances by 8 per step and MR | 8, so ni strictly grows).
//
// VGPR layouts per CDNA5 ISA 7.12.2 (wave32):
//   A 16x32 f16 : lane L (row M=L&15):   halves 0-7  = K[k0 + 8*(L>>4) .. +7]
//                                        halves 8-15 = K[k0+16+8*(L>>4) .. +7]
//   B 32x16 f16 : lane L (col N=ni*16+(L&15)): halves 0-15 = same K order as A
//   C/D 16x16   : elem e, lane L -> row M = e + 8*(L>>4), col N = L&15
template <int S, int KP, int COUT, bool FINAL>
__device__ __forceinline__ void mlp_layer(
    const _Float16* __restrict__ Xin,
    const _Float16* __restrict__ Wt, const float* __restrict__ bias,
    _Float16* __restrict__ Xout, float* __restrict__ pmax)
{
  constexpr int MR    = S / 16;
  constexpr int NC    = COUT / 16;
  constexpr int TILES = MR * NC;
  static_assert(TILES % 8 == 0, "tiles must fill 8 waves evenly");
  static_assert(8 % MR == 0, "ni must advance every tile step");

  const int tid  = threadIdx.x;
  const int lane = tid & 31;
  const int wid  = tid >> 5;
  const int row  = lane & 15;
  const int hi   = lane >> 4;

#pragma unroll
  for (int t0 = 0; t0 < TILES; t0 += 8) {
    const int t  = t0 + wid;
    const int mi = t % MR;
    const int ni = t / MR;
    v8f acc = {};
    const _Float16* rowp = Xin + (mi * 16 + row) * KP;
#pragma unroll
    for (int k0 = 0; k0 < KP; k0 += 32) {
      v8h alo = *(const v8h*)(rowp + k0 + hi * 8);
      v8h ahi = *(const v8h*)(rowp + k0 + 16 + hi * 8);
      v16h a  = __builtin_shufflevector(alo, ahi, 0, 1, 2, 3, 4, 5, 6, 7,
                                                  8, 9, 10, 11, 12, 13, 14, 15);
      v16h bf;
#pragma unroll
      for (int s = 0; s < 16; ++s)
        bf[s] = Wt[(k0 + (s < 8 ? hi * 8 + s : 16 + hi * 8 + (s - 8))) * COUT + ni * 16 + row];
      acc = __builtin_amdgcn_wmma_f32_16x16x32_f16(
          /*neg_a=*/false, a, /*neg_b=*/false, bf,
          /*c_mod=*/(short)0, acc, /*reuse_a=*/false, /*reuse_b=*/false);
    }
    const int   ccol = ni * 16 + row;
    const float bb   = bias[ccol];
    if constexpr (!FINAL) {
#pragma unroll
      for (int e = 0; e < 8; ++e) {
        float v = fmaxf(acc[e] + bb, 0.0f);                // bias + relu
        Xout[(mi * 16 + hi * 8 + e) * COUT + ccol] = (_Float16)v;
      }
    } else {
      float vmax = 0.0f;                                   // relu floor
#pragma unroll
      for (int e = 0; e < 8; ++e) vmax = fmaxf(vmax, acc[e] + bb);
      vmax = fmaxf(vmax, __shfl_xor(vmax, 16, 32));        // join half-lanes
      if (lane < 16) pmax[wid * COUT + ccol] = vmax;       // per-wave slot
    }
  }
}

// ====================== Fused ball-query + group + MLP + max ===============
// One block (256 threads) per centroid group (b,m); grid = B*M.
template <int S, int KP0, int C1, int C2, int C3>
__global__ __launch_bounds__(256) void group_mlp_kernel(
    const float* __restrict__ pts,    // (B,3,N)
    const float* __restrict__ feats,  // (B,C,N)
    const float* __restrict__ cent,   // (B,M,3) scratch
    const _Float16* __restrict__ wt0, const _Float16* __restrict__ wt1,
    const _Float16* __restrict__ wt2,
    const float* __restrict__ b0, const float* __restrict__ b1,
    const float* __restrict__ b2,
    float r2, float* __restrict__ out_feat, int ch_off)
{
  constexpr int XA = S * (KP0 > C2 ? KP0 : C2);
  __shared__ __align__(32) _Float16 Xa[XA];      // L0 in (stride KP0) / L1 out (stride C2)
  __shared__ __align__(32) _Float16 Xb[S * C1];  // L0 out / L1 in (stride C1)
  __shared__ float         pmax[8 * C3];         // per-wave column maxima
  __shared__ unsigned char flags[NPTS];
  __shared__ int           nbr[S];

  const int tid = threadIdx.x;
  const int b = blockIdx.x / MCENT;
  const int m = blockIdx.x % MCENT;

  const float* px = pts + (size_t)b * 3 * NPTS;
  const float* py = px + NPTS;
  const float* pz = px + 2 * NPTS;
  const float* cc = cent + ((size_t)b * MCENT + m) * 3;
  const float cx = cc[0], cy = cc[1], cz = cc[2];

  // warm the weight region while we compute flags (global_prefetch_b8)
  __builtin_prefetch(wt0 + (tid << 4), 0, 1);

  // ---- in-ball flags (parallel) ----
  for (int j = tid; j < NPTS; j += 256) {
    float dx = px[j] - cx, dy = py[j] - cy, dz = pz[j] - cz;
    float d2 = __fadd_rn(__fadd_rn(__fmul_rn(dx, dx), __fmul_rn(dy, dy)),
                         __fmul_rn(dz, dz));
    flags[j] = (d2 <= r2) ? 1 : 0;
  }
  for (int c = tid; c < 8 * C3; c += 256) pmax[c] = 0.0f;
  __syncthreads();

  // ---- ordered scan: first S in-ball indices, pad with first hit ----
  // (matches reference sort(where(d2<=r2, idx, n))[:S] semantics; centroid
  //  itself always qualifies with d2 == 0, so cnt >= 1)
  if (tid == 0) {
    int cnt = 0;
    for (int j = 0; j < NPTS; ++j) {
      if (flags[j]) { nbr[cnt++] = j; if (cnt == S) break; }
    }
    int f0 = (cnt > 0) ? nbr[0] : 0;
    for (; cnt < S; ++cnt) nbr[cnt] = f0;
  }
  __syncthreads();

  // ---- gather [rel_xyz ; feat] into Xa (S x KP0 f16, cols>=67 zero) ----
  for (int e = tid; e < S * KP0; e += 256) {
    int rr = e / KP0, col = e - rr * KP0;
    int j = nbr[rr];
    float v;
    if (col < 3)
      v = (col == 0) ? (px[j] - cx) : ((col == 1) ? (py[j] - cy) : (pz[j] - cz));
    else if (col < 3 + CFEAT)
      v = feats[((size_t)b * CFEAT + (col - 3)) * NPTS + j];
    else
      v = 0.0f;
    Xa[e] = (_Float16)v;
  }
  __syncthreads();

  // ---- 3 WMMA layers, ping-pong LDS; final layer folds max into pmax ----
  mlp_layer<S, KP0, C1, false>(Xa, wt0, b0, Xb, nullptr);
  __syncthreads();
  mlp_layer<S, C1, C2, false>(Xb, wt1, b1, Xa, nullptr);
  __syncthreads();
  mlp_layer<S, C2, C3, true>(Xa, wt2, b2, nullptr, pmax);
  __syncthreads();

  // ---- combine 8 per-wave slots, write (B,384,M) slice ----
  for (int c = tid; c < C3; c += 256) {
    float v = 0.0f;
#pragma unroll
    for (int w = 0; w < 8; ++w) v = fmaxf(v, pmax[w * C3 + c]);
    out_feat[((size_t)b * 384 + ch_off + c) * MCENT + m] = v;
  }
}

// ============================ launch =======================================
extern "C" void kernel_launch(void* const* d_in, const int* in_sizes, int n_in,
                              void* d_out, int out_size, void* d_ws,
                              size_t ws_size, hipStream_t stream)
{
  (void)in_sizes; (void)n_in; (void)out_size; (void)ws_size;

  const float* pts   = (const float*)d_in[0];
  const float* feats = (const float*)d_in[1];
  const float* W[6]  = {(const float*)d_in[2],  (const float*)d_in[4],
                        (const float*)d_in[6],  (const float*)d_in[8],
                        (const float*)d_in[10], (const float*)d_in[12]};
  const float* Bs[6] = {(const float*)d_in[3],  (const float*)d_in[5],
                        (const float*)d_in[7],  (const float*)d_in[9],
                        (const float*)d_in[11], (const float*)d_in[13]};

  float* out      = (float*)d_out;
  float* out_cent = out;                       // (B,3,M)
  float* out_feat = out + BATCH * 3 * MCENT;   // (B,384,M)

  // scratch layout (all 32B aligned)
  char*     ws    = (char*)d_ws;
  float*    cent  = (float*)ws;                      // B*M*3 f32 = 24576 B
  _Float16* wt0_0 = (_Float16*)(ws + 24576);         // 96 x 64
  _Float16* wt0_1 = wt0_0 + 96 * 64;                 // 64 x 64
  _Float16* wt0_2 = wt0_1 + 64 * 64;                 // 64 x 128
  _Float16* wt1_0 = wt0_2 + 64 * 128;                // 96 x 128
  _Float16* wt1_1 = wt1_0 + 96 * 128;                // 128 x 128
  _Float16* wt1_2 = wt1_1 + 128 * 128;               // 128 x 256

  fps_kernel<<<BATCH, 256, 0, stream>>>(pts, cent, out_cent);

  struct PW { const float* W; _Float16* Wt; int cout, cin, kp; };
  const PW pw[6] = {
      {W[0], wt0_0, 64, 67, 96},  {W[1], wt0_1, 64, 64, 64},
      {W[2], wt0_2, 128, 64, 64}, {W[3], wt1_0, 128, 67, 96},
      {W[4], wt1_1, 128, 128, 128}, {W[5], wt1_2, 256, 128, 128}};
  for (int i = 0; i < 6; ++i) {
    int n = pw[i].kp * pw[i].cout;
    prep_w_kernel<<<(n + 255) / 256, 256, 0, stream>>>(
        pw[i].W, pw[i].Wt, pw[i].cout, pw[i].cin, pw[i].kp);
  }

  // scale 0: S=32, r^2 = float(0.2*0.2) = 0.04f, channels [0,128)
  group_mlp_kernel<32, 96, 64, 64, 128><<<BATCH * MCENT, 256, 0, stream>>>(
      pts, feats, cent, wt0_0, wt0_1, wt0_2, Bs[0], Bs[1], Bs[2],
      0.04f, out_feat, 0);
  // scale 1: S=64, r^2 = float(0.4*0.4) = 0.16f, channels [128,384)
  group_mlp_kernel<64, 96, 128, 128, 256><<<BATCH * MCENT, 256, 0, stream>>>(
      pts, feats, cent, wt1_0, wt1_1, wt1_2, Bs[3], Bs[4], Bs[5],
      0.16f, out_feat, 128);
}
